// DIDGraphNeuralNetwork_44959717655197
// MI455X (gfx1250) — matrix operand order repaired and checked by reading, not compiled, not run on hardware
//
#include <hip/hip_runtime.h>
#include <hip/hip_bf16.h>

#define D  128
#define NH 8
#define DH 16

typedef _Float16 v16h __attribute__((ext_vector_type(16)));
typedef _Float16 v8h  __attribute__((ext_vector_type(8)));
typedef float    v8f  __attribute__((ext_vector_type(8)));

// ---------------------------------------------------------------------------
// Pack f32 [128][128] weight (row-major, W[k][n]) into WMMA B-fragment order:
// out[((ct*4+kt)*32 + lane)*16 + e] = (f16)W[(kt*32 + (lane>>4)*16 + e)*128 + ct*16 + (lane&15)]
// so each lane's 16-half fragment is one contiguous 32-byte chunk.
// ---------------------------------------------------------------------------
__global__ void k_pack_b_f16(const float* __restrict__ W, _Float16* __restrict__ out) {
  int t = blockIdx.x * blockDim.x + threadIdx.x;
  if (t >= D * D) return;
  int e    =  t        & 15;
  int lane = (t >> 4)  & 31;
  int kt   = (t >> 9)  & 3;
  int ct   =  t >> 11;
  int k = kt * 32 + (lane >> 4) * 16 + e;
  int n = ct * 16 + (lane & 15);
  out[t] = (_Float16)W[(size_t)k * D + n];
}

// ---------------------------------------------------------------------------
// WMMA GEMM: C[M,128] = A16[M,128] @ W + bias[128]
// Bp = weight pre-packed into B-fragment order (16384 halfs = 32 KB).
// Per block: async-DMA Bp into LDS once, then one wave -> one 16x128 tile.
// ---------------------------------------------------------------------------
__global__ __launch_bounds__(256)
void k_gemm128_wmma(const _Float16* __restrict__ A16,
                    const _Float16* __restrict__ Bp,
                    const float* __restrict__ bias,   // always valid (zeros if none)
                    float* __restrict__ C, int M)
{
  __shared__ _Float16 sB[D * D];                       // 32 KB

  // cooperative async global->LDS staging: 8 x b128 per thread
  {
    const unsigned ldsBase = (unsigned)(uintptr_t)sB;
    const unsigned toff = threadIdx.x * 16;            // 16B chunk per thread
#pragma unroll
    for (int i = 0; i < 8; ++i) {
      unsigned lds_addr = ldsBase + toff + i * 4096;   // 256 thr * 16B stride
      unsigned goff     = toff + i * 4096;
      asm volatile("global_load_async_to_lds_b128 %0, %1, %2"
                   :: "v"(lds_addr), "v"(goff), "s"(Bp) : "memory");
    }
    asm volatile("s_wait_asynccnt 0x0" ::: "memory");
  }
  __syncthreads();

  // scalarize the per-wave tile index: SGPR branches, EXEC never touched
  const int tile = __builtin_amdgcn_readfirstlane(blockIdx.x * 8 + (threadIdx.x >> 5));
  const int row0 = tile << 4;
  if (row0 >= M) return;                  // scalar branch
  const int lane = threadIdx.x & 31;
  const int m    = lane & 15;
  const int g    = lane >> 4;
  const bool full = (row0 + 16 <= M);     // scalar

  int arow = row0 + m; if (arow >= M) arow = M - 1;
  const _Float16* Ar = A16 + (size_t)arow * D;

  // A fragment, 16-bit 16x32 layout: per lane two contiguous 8-half chunks.
  v16h a[4];
#pragma unroll
  for (int kt = 0; kt < 4; ++kt) {
    v8h lo = *(const v8h*)(Ar + kt * 32 + g * 16);
    v8h hi = *(const v8h*)(Ar + kt * 32 + g * 16 + 8);
#pragma unroll
    for (int i = 0; i < 8; ++i) { a[kt][i] = lo[i]; a[kt][8 + i] = hi[i]; }
  }

  const v16h* sBf = (const v16h*)sB;      // 32-byte fragments
  const int n = lane & 15;
  for (int ct = 0; ct < 8; ++ct) {
    const int ncol = ct * 16 + n;
    v8f acc = {};
#pragma unroll
    for (int kt = 0; kt < 4; ++kt) {
      v16h b = sBf[(ct * 4 + kt) * 32 + lane];
      acc = __builtin_amdgcn_wmma_f32_16x16x32_f16(false, a[kt], false, b,
                                                   (short)0, acc, false, false);
    }
    const float bv = bias[ncol];
    float* Cp = C + (size_t)(row0 + g * 8) * D + ncol;   // C/D layout: row = g*8 + r
    if (full) {                                           // scalar branch
#pragma unroll
      for (int r = 0; r < 8; ++r) Cp[(size_t)r * D] = acc[r] + bv;
    } else {
#pragma unroll
      for (int r = 0; r < 8; ++r) {
        const int rr = row0 + g * 8 + r;
        if (rr < M) C[(size_t)rr * D + ncol] = acc[r] + bv;
      }
    }
  }
}

// ---------------------------------------------------------------------------
// Small utility kernels
// ---------------------------------------------------------------------------
__global__ void k_cvt_f16(const float* __restrict__ s, _Float16* __restrict__ d, int n) {
  int i = blockIdx.x * blockDim.x + threadIdx.x;
  if (i < n) d[i] = (_Float16)s[i];
}
__global__ void k_fill_f32(float* p, float v, int n) {
  int i = blockIdx.x * blockDim.x + threadIdx.x;
  if (i < n) p[i] = v;
}
__global__ void k_fill_u32(unsigned* p, unsigned v, int n) {
  int i = blockIdx.x * blockDim.x + threadIdx.x;
  if (i < n) p[i] = v;
}
__global__ void k_copy_f32(const float* __restrict__ s, float* __restrict__ d, int n) {
  int i = blockIdx.x * blockDim.x + threadIdx.x;
  if (i < n) d[i] = s[i];
}
__global__ void k_add2(const float* __restrict__ a, const float* __restrict__ b,
                       float* __restrict__ o, int n) {
  int i = blockIdx.x * blockDim.x + threadIdx.x;
  if (i < n) o[i] = a[i] + b[i];
}
__global__ void k_relu_add2(const float* __restrict__ a, const float* __restrict__ b,
                            float* __restrict__ o, int n) {
  int i = blockIdx.x * blockDim.x + threadIdx.x;
  if (i < n) o[i] = fmaxf(a[i] + b[i], 0.0f);
}

// h[i] = emb[x[i]]  (N*32 threads, float4 each)
__global__ void k_embed(const int* __restrict__ x, const float* __restrict__ emb,
                        float* __restrict__ h, int N) {
  int t = blockIdx.x * blockDim.x + threadIdx.x;
  int i = t >> 5, j = (t & 31) << 2;
  if (i < N)
    *(float4*)(h + (size_t)i * D + j) = *(const float4*)(emb + (size_t)x[i] * D + j);
}

// ordered-uint encoding for float atomic max
__device__ __forceinline__ unsigned f2ord(float f) {
  unsigned u = __float_as_uint(f);
  return (u & 0x80000000u) ? ~u : (u | 0x80000000u);
}
__device__ __forceinline__ float ord2f(unsigned u) {
  return (u & 0x80000000u) ? __uint_as_float(u & 0x7FFFFFFFu) : __uint_as_float(~u);
}

// logits[e,h] = (Q[col[e],h,:] . K[row[e],h,:]) / 4 ; segmax = per-(col,h) max
__global__ void k_edge_logits(const int* __restrict__ row, const int* __restrict__ col,
                              const float* __restrict__ Q, const float* __restrict__ K,
                              float* __restrict__ logits, unsigned* __restrict__ segmax, int E) {
  int t = blockIdx.x * blockDim.x + threadIdx.x;
  if (t >= E * NH) return;
  int e = t >> 3, hd = t & 7;
  int r = row[e], c = col[e];
  const float* q = Q + (size_t)c * D + hd * DH;
  const float* k = K + (size_t)r * D + hd * DH;
  float s = 0.0f;
#pragma unroll
  for (int j = 0; j < DH; j += 4) {
    float4 qv = *(const float4*)(q + j), kv = *(const float4*)(k + j);
    s += qv.x * kv.x + qv.y * kv.y + qv.z * kv.z + qv.w * kv.w;
  }
  s *= 0.25f;                 // 1/sqrt(16)
  logits[t] = s;
  atomicMax(&segmax[(size_t)c * NH + hd], f2ord(s));
}

__global__ void k_edge_exp(const int* __restrict__ col, float* __restrict__ logits,
                           const unsigned* __restrict__ segmax, float* __restrict__ den, int E) {
  int t = blockIdx.x * blockDim.x + threadIdx.x;
  if (t >= E * NH) return;
  int e = t >> 3, hd = t & 7;
  int c = col[e];
  float ex = __expf(logits[t] - ord2f(segmax[(size_t)c * NH + hd]));
  logits[t] = ex;
  atomicAdd(&den[(size_t)c * NH + hd], ex);
}

__global__ void k_edge_agg(const int* __restrict__ row, const int* __restrict__ col,
                           const float* __restrict__ ex, const float* __restrict__ den,
                           const float* __restrict__ V, float* __restrict__ agg, int E) {
  int t = blockIdx.x * blockDim.x + threadIdx.x;
  if (t >= E * NH) return;
  int e = t >> 3, hd = t & 7;
  int r = row[e], c = col[e];
  float att = ex[t] / (den[(size_t)c * NH + hd] + 1e-16f);
  const float* v = V + (size_t)r * D + hd * DH;
  float* o = agg + (size_t)c * D + hd * DH;
#pragma unroll
  for (int j = 0; j < DH; ++j) atomicAdd(o + j, att * v[j]);
}

__global__ void k_edge_count(const int* __restrict__ col, float* __restrict__ cnt, int E) {
  int e = blockIdx.x * blockDim.x + threadIdx.x;
  if (e < E) atomicAdd(&cnt[col[e]], 1.0f);
}
__global__ void k_dinv(const float* __restrict__ cnt, float* __restrict__ dinv, int N) {
  int i = blockIdx.x * blockDim.x + threadIdx.x;
  if (i < N) dinv[i] = rsqrtf(cnt[i] + 1.0f);
}

// gcn[col] += dinv[row]*dinv[col] * xw[row]   (E*32 threads, 4 floats each)
__global__ void k_edge_gcn(const int* __restrict__ row, const int* __restrict__ col,
                           const float* __restrict__ dinv, const float* __restrict__ xw,
                           float* __restrict__ gcn, int E) {
  int t = blockIdx.x * blockDim.x + threadIdx.x;
  int e = t >> 5, j = (t & 31) << 2;
  if (e >= E) return;
  int r = row[e], c = col[e];
  float nm = dinv[r] * dinv[c];
  float4 xv = *(const float4*)(xw + (size_t)r * D + j);
  float* o = gcn + (size_t)c * D + j;
  atomicAdd(o + 0, nm * xv.x); atomicAdd(o + 1, nm * xv.y);
  atomicAdd(o + 2, nm * xv.z); atomicAdd(o + 3, nm * xv.w);
}

__global__ void k_gcn_fin(const float* __restrict__ gcn, const float* __restrict__ xw,
                          const float* __restrict__ cnt, const float* __restrict__ bg,
                          float* __restrict__ h2, int N) {
  int t = blockIdx.x * blockDim.x + threadIdx.x;
  if (t >= N * D) return;
  int i = t >> 7, j = t & 127;
  float deg = cnt[i] + 1.0f;
  h2[t] = fmaxf(gcn[t] + xw[t] / deg + bg[j], 0.0f);
}

// sum[col] += h[row]   (E*32 threads, 4 floats each)
__global__ void k_edge_sum(const int* __restrict__ row, const int* __restrict__ col,
                           const float* __restrict__ h, float* __restrict__ sum, int E) {
  int t = blockIdx.x * blockDim.x + threadIdx.x;
  int e = t >> 5, j = (t & 31) << 2;
  if (e >= E) return;
  float4 hv = *(const float4*)(h + (size_t)row[e] * D + j);
  float* o = sum + (size_t)col[e] * D + j;
  atomicAdd(o + 0, hv.x); atomicAdd(o + 1, hv.y);
  atomicAdd(o + 2, hv.z); atomicAdd(o + 3, hv.w);
}
__global__ void k_mean(float* __restrict__ sum, const float* __restrict__ cnt, int N) {
  int t = blockIdx.x * blockDim.x + threadIdx.x;
  if (t >= N * D) return;
  sum[t] /= fmaxf(cnt[t >> 7], 1.0f);
}

// trust edge weights: ew = softplus([h[row],h[col]] . We + be); rs[row] += ew
__global__ void k_edge_ew(const int* __restrict__ row, const int* __restrict__ col,
                          const float* __restrict__ h, const float* __restrict__ We,
                          const float* __restrict__ be, float* __restrict__ ew,
                          float* __restrict__ rs, int E) {
  int e = blockIdx.x * blockDim.x + threadIdx.x;
  if (e >= E) return;
  const float* hr = h + (size_t)row[e] * D;
  const float* hc = h + (size_t)col[e] * D;
  float s = be[0];
#pragma unroll 4
  for (int j = 0; j < D; j += 4) {
    float4 a = *(const float4*)(hr + j), wa = *(const float4*)(We + j);
    float4 b = *(const float4*)(hc + j), wb = *(const float4*)(We + D + j);
    s += a.x * wa.x + a.y * wa.y + a.z * wa.z + a.w * wa.w;
    s += b.x * wb.x + b.y * wb.y + b.z * wb.z + b.w * wb.w;
  }
  float sp = (s > 20.0f) ? s : log1pf(__expf(s));
  ew[e] = sp;
  atomicAdd(&rs[row[e]], sp);
}
__global__ void k_edge_wn(const int* __restrict__ row, const float* __restrict__ ew,
                          const float* __restrict__ rs, float* __restrict__ wn, int E) {
  int e = blockIdx.x * blockDim.x + threadIdx.x;
  if (e >= E) return;
  float r = rs[row[e]];
  wn[e] = ew[e] * 0.85f / ((r == 0.0f) ? 1.0f : r);   // alpha folded in
}
__global__ void k_edge_diffuse(const int* __restrict__ row, const int* __restrict__ col,
                               const float* __restrict__ wn, const float* __restrict__ tin,
                               float* __restrict__ tout, int E) {
  int e = blockIdx.x * blockDim.x + threadIdx.x;
  if (e >= E) return;
  atomicAdd(&tout[row[e]], wn[e] * tin[col[e]]);
}

// anomaly head
__global__ void k_anom1(const float* __restrict__ h, const float* __restrict__ Wa1,
                        const float* __restrict__ ba1, float* __restrict__ a1, int N) {
  int t = blockIdx.x * blockDim.x + threadIdx.x;
  if (t >= N * 64) return;
  int i = t >> 6, j = t & 63;
  float s = ba1[j];
  const float* hr = h + (size_t)i * D;
#pragma unroll 8
  for (int k = 0; k < D; ++k) s += hr[k] * Wa1[(size_t)k * 64 + j];
  a1[t] = fmaxf(s, 0.0f);
}
__global__ void k_anom2(const float* __restrict__ a1, const float* __restrict__ Wa2,
                        const float* __restrict__ ba2, float* __restrict__ out, int N) {
  int i = blockIdx.x * blockDim.x + threadIdx.x;
  if (i >= N) return;
  float s = ba2[0];
  const float* a = a1 + (size_t)i * 64;
#pragma unroll 8
  for (int k = 0; k < 64; ++k) s += a[k] * Wa2[k];
  out[i] = 1.0f / (1.0f + __expf(-s));
}

// ---------------------------------------------------------------------------
extern "C" void kernel_launch(void* const* d_in, const int* in_sizes, int n_in,
                              void* d_out, int out_size, void* d_ws, size_t ws_size,
                              hipStream_t stream)
{
  const int N = in_sizes[0];
  const int E = in_sizes[1] / 2;

  const int*   x    = (const int*)d_in[0];
  const int*   row  = (const int*)d_in[1];
  const int*   col  = row + E;
  const float* emb  = (const float*)d_in[2];
  const float* Wq = (const float*)d_in[3],  *bq = (const float*)d_in[4];
  const float* Wk = (const float*)d_in[5],  *bk = (const float*)d_in[6];
  const float* Wv = (const float*)d_in[7],  *bv = (const float*)d_in[8];
  const float* Ws = (const float*)d_in[9],  *bs = (const float*)d_in[10];
  const float* Wg = (const float*)d_in[11], *bg = (const float*)d_in[12];
  const float* Wl = (const float*)d_in[13], *bl = (const float*)d_in[14];
  const float* Wr = (const float*)d_in[15];
  const float* We = (const float*)d_in[16], *be = (const float*)d_in[17];
  const float* Wa1 = (const float*)d_in[18], *ba1 = (const float*)d_in[19];
  const float* Wa2 = (const float*)d_in[20], *ba2 = (const float*)d_in[21];

  // deterministic workspace carve-out
  char* ws = (char*)d_ws;
  size_t off = 0;
  auto carve = [&](size_t bytes) -> char* {
    char* p = ws + off;
    off = (off + bytes + 255) & ~(size_t)255;
    return p;
  };
  _Float16* w16   = (_Float16*)carve((size_t)7 * D * D * sizeof(_Float16));
  _Float16* h16   = (_Float16*)carve((size_t)N * D * sizeof(_Float16));
  float*    zbias = (float*)carve((size_t)D * 4);       // zero bias for no-bias GEMMs
  float*    hbuf  = (float*)carve((size_t)N * D * 4);
  float*    bufA  = (float*)carve((size_t)N * D * 4);   // Q / agg / sum-mean / a1
  float*    bufB  = (float*)carve((size_t)N * D * 4);   // K / g2
  float*    bufC  = (float*)carve((size_t)N * D * 4);   // V / xw / g1
  float*    bufD  = (float*)carve((size_t)N * D * 4);   // S / gcn
  float*    logits = (float*)carve((size_t)E * NH * 4); // later holds exp()
  unsigned* segmax = (unsigned*)carve((size_t)N * NH * 4);
  float*    den   = (float*)carve((size_t)N * NH * 4);
  float*    cnt   = (float*)carve((size_t)N * 4);
  float*    dinv  = (float*)carve((size_t)N * 4);
  float*    ewb   = (float*)carve((size_t)E * 4);
  float*    rs    = (float*)carve((size_t)N * 4);
  float*    wnb   = (float*)carve((size_t)E * 4);
  float*    tA    = (float*)carve((size_t)N * 4);
  float*    tB    = (float*)carve((size_t)N * 4);

  float* hOut = (float*)d_out;              // final h lives in d_out[0 : N*D)
  float* tOut = hOut + (size_t)N * D;       // trust
  float* aOut = tOut + N;                   // anomaly

  const int B256 = 256;
  auto g = [](long n) { return (unsigned)((n + 255) / 256); };
  const unsigned gemmBlocks = (unsigned)(((N + 15) / 16 + 7) / 8);

  // 0) pack weights into WMMA B-fragment f16 order (Wq,Wk,Wv,Wskip,Wg,Wl,Wr)
  const float* Wmats[7] = {Wq, Wk, Wv, Ws, Wg, Wl, Wr};
  for (int i = 0; i < 7; ++i)
    k_pack_b_f16<<<g(D * D), B256, 0, stream>>>(Wmats[i], w16 + (size_t)i * D * D);
  k_fill_f32<<<1, B256, 0, stream>>>(zbias, 0.0f, D);

  // 1) embedding lookup -> h0
  k_embed<<<g((long)N * 32), B256, 0, stream>>>(x, emb, hbuf, N);

  // 2) layer 0: four WMMA GEMMs sharing A = h0
  k_cvt_f16<<<g((long)N * D), B256, 0, stream>>>(hbuf, h16, N * D);
  k_gemm128_wmma<<<gemmBlocks, B256, 0, stream>>>(h16, w16 + 0 * D * D, bq, bufA, N); // Q
  k_gemm128_wmma<<<gemmBlocks, B256, 0, stream>>>(h16, w16 + 1 * D * D, bk, bufB, N); // K
  k_gemm128_wmma<<<gemmBlocks, B256, 0, stream>>>(h16, w16 + 2 * D * D, bv, bufC, N); // V
  k_gemm128_wmma<<<gemmBlocks, B256, 0, stream>>>(h16, w16 + 3 * D * D, bs, bufD, N); // skip

  // 3) segmented softmax attention over edges
  k_fill_u32<<<g((long)N * NH), B256, 0, stream>>>(segmax, 0u, N * NH);
  k_fill_f32<<<g((long)N * NH), B256, 0, stream>>>(den, 0.0f, N * NH);
  k_edge_logits<<<g((long)E * NH), B256, 0, stream>>>(row, col, bufA, bufB, logits, segmax, E);
  k_edge_exp<<<g((long)E * NH), B256, 0, stream>>>(col, logits, segmax, den, E);
  k_fill_f32<<<g((long)N * D), B256, 0, stream>>>(bufA, 0.0f, N * D);                 // agg
  k_edge_agg<<<g((long)E * NH), B256, 0, stream>>>(row, col, logits, den, bufC, bufA, E);
  k_relu_add2<<<g((long)N * D), B256, 0, stream>>>(bufA, bufD, hbuf, N * D);          // h1

  // 4) layer 1: GCN
  k_cvt_f16<<<g((long)N * D), B256, 0, stream>>>(hbuf, h16, N * D);
  k_gemm128_wmma<<<gemmBlocks, B256, 0, stream>>>(h16, w16 + 4 * D * D, zbias, bufC, N); // xw
  k_fill_f32<<<g(N), B256, 0, stream>>>(cnt, 0.0f, N);
  k_edge_count<<<g(E), B256, 0, stream>>>(col, cnt, E);
  k_dinv<<<g(N), B256, 0, stream>>>(cnt, dinv, N);
  k_fill_f32<<<g((long)N * D), B256, 0, stream>>>(bufD, 0.0f, N * D);                 // gcn
  k_edge_gcn<<<g((long)E * 32), B256, 0, stream>>>(row, col, dinv, bufC, bufD, E);
  k_gcn_fin<<<g((long)N * D), B256, 0, stream>>>(bufD, bufC, cnt, bg, hbuf, N);       // h2

  // 5) layer 2: SAGE (mean-agg)
  k_fill_f32<<<g((long)N * D), B256, 0, stream>>>(bufA, 0.0f, N * D);                 // sum
  k_edge_sum<<<g((long)E * 32), B256, 0, stream>>>(row, col, hbuf, bufA, E);
  k_mean<<<g((long)N * D), B256, 0, stream>>>(bufA, cnt, N);
  k_cvt_f16<<<g((long)N * D), B256, 0, stream>>>(hbuf, h16, N * D);
  k_gemm128_wmma<<<gemmBlocks, B256, 0, stream>>>(h16, w16 + 6 * D * D, zbias, bufB, N); // h2@Wr
  k_cvt_f16<<<g((long)N * D), B256, 0, stream>>>(bufA, h16, N * D);
  k_gemm128_wmma<<<gemmBlocks, B256, 0, stream>>>(h16, w16 + 5 * D * D, bl, bufC, N); // mean@Wl+bl
  k_add2<<<g((long)N * D), B256, 0, stream>>>(bufB, bufC, hOut, N * D);               // final h

  // 6) trust propagation (sparse row-normalized SpMV, 5 steps)
  k_fill_f32<<<g(N), B256, 0, stream>>>(rs, 0.0f, N);
  k_edge_ew<<<g(E), B256, 0, stream>>>(row, col, hOut, We, be, ewb, rs, E);
  k_edge_wn<<<g(E), B256, 0, stream>>>(row, ewb, rs, wnb, E);
  k_fill_f32<<<g(N), B256, 0, stream>>>(tA, 0.5f, N);
  float* tin = tA; float* tout = tB;
  for (int it = 0; it < 5; ++it) {
    k_fill_f32<<<g(N), B256, 0, stream>>>(tout, 0.075f, N);   // (1-alpha)*0.5
    k_edge_diffuse<<<g(E), B256, 0, stream>>>(row, col, wnb, tin, tout, E);
    float* tmp = tin; tin = tout; tout = tmp;
  }
  k_copy_f32<<<g(N), B256, 0, stream>>>(tin, tOut, N);

  // 7) anomaly head
  k_anom1<<<g((long)N * 64), B256, 0, stream>>>(hOut, Wa1, ba1, bufA, N);
  k_anom2<<<g(N), B256, 0, stream>>>(bufA, Wa2, ba2, aOut, N);

  (void)n_in; (void)out_size; (void)ws_size;
}
